// Seq2SeqAttention_12584254178079
// MI455X (gfx1250) — matrix operand order repaired and checked by reading, not compiled, not run on hardware
//
#include <hip/hip_runtime.h>
#include <hip/hip_bf16.h>

// ---------------------------------------------------------------------------
// Seq2Seq attention RNN, persistent-kernel design for MI455X (gfx1250).
//   * bf16 WMMA 16x16x32 for all GEMMs, fp32 accumulate
//   * 8 workgroups x 8 wave32 = 64 waves == tile grid of every per-step GEMM
//   * k-outer / gate-inner WMMA ordering: 4 independent accumulators,
//     A-fragment reused 4x, no WMMA RAW chains
//   * all global traffic through explicit addrspace(1) -> global_load/store
//     (not flat), weights L2-resident bf16
//   * activations staged to LDS via async-to-LDS DMA (ASYNCcnt)
//   * cluster barrier (CDNA5) for cross-WGP step synchronization
// ---------------------------------------------------------------------------

typedef __bf16 bf16_t;
typedef __attribute__((ext_vector_type(16))) __bf16 v16bf;
typedef __attribute__((ext_vector_type(8)))  float  v8f;
typedef __attribute__((ext_vector_type(4)))  int    v4i;

#define BATCH    64
#define T_STEPS  400
#define F_IN     80
#define F_PAD    96      // 80 padded to multiple of 32 (K of WMMA)
#define HID      256
#define EMB      256
#define CTXW     20
#define NWG      8
#define THREADS  256
#define LDS_H_OFF (64*512)          // second staging region (h input, K<=256)
#define LDS_BYTES (64*768*2)        // 96 KB dynamic LDS (sX up to 64x512)

#if __has_builtin(__builtin_amdgcn_global_load_async_to_lds_b128)
#define HAS_ASYNC_LDS 1
#else
#define HAS_ASYNC_LDS 0
#endif

// explicit address-space pointer types: force global_* (not flat_*) memops
typedef __attribute__((address_space(1))) v4i          g_v4i;
typedef __attribute__((address_space(1))) const v4i    gc_v4i;
typedef __attribute__((address_space(3))) v4i          l_v4i;
typedef __attribute__((address_space(1))) bf16_t       g_bf16;
typedef __attribute__((address_space(1))) const bf16_t gc_bf16;
typedef __attribute__((address_space(1))) float        g_float;
typedef __attribute__((address_space(1))) const float  gc_float;

// ----------------------------- device helpers ------------------------------

__device__ __forceinline__ float sigm(float x) { return 1.f / (1.f + __expf(-x)); }
__device__ __forceinline__ float tanh_(float x) {
  float e = __expf(2.f * x);
  return (e - 1.f) / (e + 1.f);
}

// A fragment from LDS: 16x32 bf16, rows striped over lanes (ISA 7.12.2)
__device__ __forceinline__ v16bf ldA(const bf16_t* a, int K, int mt, int k0, int lane) {
  const bf16_t* p = a + (size_t)(mt * 16 + (lane & 15)) * K + k0 + ((lane >> 4) << 3);
  v16bf r;
  ((uint4*)&r)[0] = *(const uint4*)(p);
  ((uint4*)&r)[1] = *(const uint4*)(p + 16);
  return r;
}
// B fragment from global: 32x16 bf16; lane = output column, K = 8h+s and
// 16+8h+(s-8) per lane (matches ldA). Weight row-major [N][K] (out[m,n] = sum_k A[m,k] * W[n,k]).
__device__ __forceinline__ v16bf ldB(gc_bf16* wrow0, int K, int k0, int lane) {
  gc_bf16* p = wrow0 + (size_t)(lane & 15) * K + k0 + ((lane >> 4) << 3);
  v16bf r;
  ((v4i*)&r)[0] = ((gc_v4i*)p)[0];
  ((v4i*)&r)[1] = ((gc_v4i*)(p + 16))[0];
  return r;
}
__device__ __forceinline__ v8f wmma_bf16(v16bf a, v16bf b, v8f c) {
  return __builtin_amdgcn_wmma_f32_16x16x32_bf16(false, a, false, b, (short)0, c,
                                                 false, false);
}

// cross-workgroup step barrier: workgroup s_barrier + CDNA5 cluster barrier
__device__ __forceinline__ void grid_sync() {
  __threadfence();
  __syncthreads();
  if (threadIdx.x < 32) __builtin_amdgcn_s_cluster_barrier();
  __syncthreads();
}

// cooperative global->LDS staging via CDNA5 async-to-LDS DMA (ASYNCcnt path,
// no VGPR round trip); falls back to b128+ds_store if builtin is absent.
// elems is always a multiple of 8*THREADS here -> no divergent tail.
__device__ __forceinline__ void stage(bf16_t* dst, const bf16_t* src, int elems, int tid) {
  const int n = elems >> 3;                      // 16B packets
#if HAS_ASYNC_LDS
  __attribute__((address_space(1))) v4i* g =
      (__attribute__((address_space(1))) v4i*)(const_cast<bf16_t*>(src));
  l_v4i* l = (l_v4i*)(dst);
  for (int i = tid; i < n; i += THREADS)
    __builtin_amdgcn_global_load_async_to_lds_b128(g + i, l + i, 0, 0);
#else
  uint4* d = (uint4*)dst;
  const uint4* s = (const uint4*)src;
  for (int i = tid; i < n; i += THREADS) d[i] = s[i];
#endif
}

// fence staged data before the workgroup barrier
__device__ __forceinline__ void stage_fence() {
#if HAS_ASYNC_LDS
#if __has_builtin(__builtin_amdgcn_s_wait_asynccnt)
  __builtin_amdgcn_s_wait_asynccnt(0);
#else
  asm volatile("s_wait_asynccnt 0x0" ::: "memory");
#endif
#endif
  __syncthreads();
}

__device__ __forceinline__ void load8f(gc_bf16* p, float* o) {
  v4i raw = *(gc_v4i*)p;
  const bf16_t* h = (const bf16_t*)&raw;
#pragma unroll
  for (int j = 0; j < 8; ++j) o[j] = (float)h[j];
}

// ------------------------------- GEMM cores --------------------------------

// Fused LSTM layer: gates = sX@Wih^T + sH@Whh^T + b, then pointwise cell.
// Wave owns (unit-tile, m-tile); k-outer/gate-inner: one A fragment feeds 4
// WMMAs into 4 independent accumulators (no RAW chain between WMMAs).
template <int KX, int KH, int NU>
__device__ void lstm_layer(const bf16_t* sX, gc_bf16* Wih,
                           const bf16_t* sH, gc_bf16* Whh,
                           gc_float* bias, g_float* C, g_bf16* Hout,
                           int hStride, g_float* outp, int outOff,
                           int slot, int lane) {
  constexpr int tiles = (NU / 16) * 4;        // (NU/16) unit tiles * 4 m tiles
  for (int s = slot; s < tiles; s += 64) {
    const int ut = s >> 2, mt = s & 3;
    const v8f zero = {0.f, 0.f, 0.f, 0.f, 0.f, 0.f, 0.f, 0.f};
    v8f acc[4] = {zero, zero, zero, zero};
    gc_bf16* wx[4];
    gc_bf16* wh[4];
#pragma unroll
    for (int g = 0; g < 4; ++g) {
      wx[g] = Wih + (size_t)(g * NU + ut * 16) * KX;
      wh[g] = Whh + (size_t)(g * NU + ut * 16) * KH;
    }
#pragma unroll
    for (int k = 0; k < KX; k += 32) {
      const v16bf a = ldA(sX, KX, mt, k, lane);
#pragma unroll
      for (int g = 0; g < 4; ++g)
        acc[g] = wmma_bf16(a, ldB(wx[g], KX, k, lane), acc[g]);
    }
#pragma unroll
    for (int k = 0; k < KH; k += 32) {
      const v16bf a = ldA(sH, KH, mt, k, lane);
#pragma unroll
      for (int g = 0; g < 4; ++g)
        acc[g] = wmma_bf16(a, ldB(wh[g], KH, k, lane), acc[g]);
    }
    const int u = ut * 16 + (lane & 15);
    const float bi = bias[u], bf_ = bias[NU + u];
    const float bg = bias[2 * NU + u], bo = bias[3 * NU + u];
    const int mbase = mt * 16 + ((lane >> 4) << 3);   // C/D layout: lanes>=16 -> M+8
#pragma unroll
    for (int r = 0; r < 8; ++r) {
      const int m = mbase + r;
      const float gi = sigm(acc[0][r] + bi);
      const float gf = sigm(acc[1][r] + bf_);
      const float gg = tanh_(acc[2][r] + bg);
      const float go = sigm(acc[3][r] + bo);
      const float cn = gf * C[(size_t)m * NU + u] + gi * gg;
      C[(size_t)m * NU + u] = cn;
      const float h = go * tanh_(cn);
      Hout[(size_t)m * hStride + u] = (bf16_t)h;
      if (outp) outp[(size_t)m * (380 * 80) + outOff + u] = h;
    }
  }
}

// Plain GEMM + bias, N == 256 (16 n-tiles x 4 m-tiles == 64 wave slots)
template <int K>
__device__ void gemm_bias(const bf16_t* sX, gc_bf16* W,
                          gc_float* bias, g_bf16* out, int outStride,
                          int slot, int lane) {
  const int nt = slot >> 2, mt = slot & 3;
  v8f acc = {0.f, 0.f, 0.f, 0.f, 0.f, 0.f, 0.f, 0.f};
  gc_bf16* wr = W + (size_t)(nt * 16) * K;
#pragma unroll
  for (int k = 0; k < K; k += 32)
    acc = wmma_bf16(ldA(sX, K, mt, k, lane), ldB(wr, K, k, lane), acc);
  const int n = nt * 16 + (lane & 15);
  const float bv = bias[n];
  const int mbase = mt * 16 + ((lane >> 4) << 3);
#pragma unroll
  for (int r = 0; r < 8; ++r)
    out[(size_t)(mbase + r) * outStride + n] = (bf16_t)(acc[r] + bv);
}

// ------------------------------- parameters --------------------------------

struct SeqParams {
  const float* x;
  const float *e1_b, *e2_b, *e3_b, *d1_b, *d2_b, *d3_b, *emb_b, *key_b, *val_b;
  const bf16_t *emb_W, *e1_Wih, *e1_Whh, *e2_Wih, *e2_Whh, *e3_Wih, *e3_Whh,
               *d1_Wih, *d1_Whh, *d2_Wih, *d2_Whh, *d3_Wih, *d3_Whh, *key_W, *val_W;
  bf16_t *xt, *xprev, *din, *ktmp, *vtmp, *kbuf, *vbuf;
  bf16_t *h_e1[2], *h_e2[2], *h_e3[2], *h_d1[2], *h_d2[2], *h_d3[2];
  float *c_e1, *c_e2, *c_e3, *c_d1, *c_d2, *c_d3;
  float* out;
};

#define GW(x)  ((gc_bf16*)(x))     // global const bf16 (weights)
#define GB(x)  ((gc_float*)(x))    // global const float (bias)
#define GF(x)  ((g_float*)(x))     // global float (cell state / out)
#define GH(x)  ((g_bf16*)(x))      // global bf16 (activations)

// ------------------------------- main kernel -------------------------------

__global__ __launch_bounds__(THREADS, 1) void seq2seq_main(SeqParams P) {
  extern __shared__ __align__(16) unsigned char smem[];
  bf16_t* lds  = (bf16_t*)smem;          // sX region (up to 64x512)
  bf16_t* ldsH = lds + LDS_H_OFF;        // sH region (up to 64x256)

  const int tid  = threadIdx.x;
  const int lane = tid & 31;
  const int wave = tid >> 5;
  const int wg   = blockIdx.x;
  const int slot = wg * 8 + wave;        // 0..63 global wave slot

  for (int t = 0; t < T_STEPS; ++t) {
    const int p = t & 1, q = p ^ 1;
    bf16_t *he1r = P.h_e1[p], *he1w = P.h_e1[q];
    bf16_t *he2r = P.h_e2[p], *he2w = P.h_e2[q];
    bf16_t *he3r = P.h_e3[p], *he3w = P.h_e3[q];
    bf16_t *hd1r = P.h_d1[p], *hd1w = P.h_d1[q];
    bf16_t *hd2r = P.h_d2[p], *hd2w = P.h_d2[q];
    bf16_t *hd3r = P.h_d3[p], *hd3w = P.h_d3[q];

    // stage-0: build bf16 encoder input from x for warmup steps (WG0 only)
    if (wg == 0 && t <= CTXW) {
      gc_float* xs = (gc_float*)P.x;
      g_bf16*   xd = GH(P.xt);
      g_bf16*   xp = GH(P.xprev);
      for (int i = tid; i < BATCH * F_IN; i += THREADS) {
        const int b = i / F_IN, f = i - b * F_IN;
        xd[b * F_PAD + f] =
            (bf16_t)xs[(size_t)b * (T_STEPS * F_IN) + (size_t)t * F_IN + f];
        if (t == CTXW)
          xp[b * F_PAD + f] =
              (bf16_t)xs[(size_t)b * (T_STEPS * F_IN) + (size_t)(CTXW - 1) * F_IN + f];
      }
    }
    grid_sync();   // prev-step h writes + xt visible cluster-wide

    // ------------------------- encoder stack -------------------------------
    const bf16_t* xin = (t <= CTXW) ? P.xt : hd3r;   // autoregressive feedback
    stage(lds,  xin,  BATCH * F_PAD, tid);
    stage(ldsH, he1r, BATCH * HID,   tid);
    stage_fence();
    lstm_layer<F_PAD, HID, HID>(lds, GW(P.e1_Wih), ldsH, GW(P.e1_Whh), GB(P.e1_b),
                                GF(P.c_e1), GH(he1w), HID, (g_float*)nullptr, 0,
                                slot, lane);
    grid_sync();

    stage(lds,  he1w, BATCH * HID, tid);
    stage(ldsH, he2r, BATCH * HID, tid);
    stage_fence();
    lstm_layer<HID, HID, HID>(lds, GW(P.e2_Wih), ldsH, GW(P.e2_Whh), GB(P.e2_b),
                              GF(P.c_e2), GH(he2w), HID, (g_float*)nullptr, 0,
                              slot, lane);
    grid_sync();

    stage(lds,  he2w, BATCH * HID, tid);
    stage(ldsH, he3r, BATCH * HID, tid);
    stage_fence();
    lstm_layer<HID, HID, HID>(lds, GW(P.e3_Wih), ldsH, GW(P.e3_Whh), GB(P.e3_b),
                              GF(P.c_e3), GH(he3w), HID, (g_float*)nullptr, 0,
                              slot, lane);
    grid_sync();

    // --------------------- key / value / embedding -------------------------
    stage(lds, he3w, BATCH * HID, tid);
    const bf16_t* embA = (t == CTXW) ? P.xprev : hd3r;
    if (t >= CTXW) stage(ldsH, embA, BATCH * F_PAD, tid);
    stage_fence();
    gemm_bias<HID>(lds, GW(P.key_W), GB(P.key_b), GH(P.ktmp), EMB, slot, lane);
    gemm_bias<HID>(lds, GW(P.val_W), GB(P.val_b), GH(P.vtmp), EMB, slot, lane);
    if (t >= CTXW)
      gemm_bias<F_PAD>(ldsH, GW(P.emb_W), GB(P.emb_b), GH(P.din), 2 * EMB,
                       slot, lane);
    grid_sync();

    // ------------------- attention + ring update (wave/batch) --------------
    {
      const int b = slot;            // 64 waves == 64 batch rows
      const int rs = t % CTXW;
      if (t > CTXW) {                // softmax attention over last-20 window
        float qv[8];
        load8f(GW(hd2r) + (size_t)b * EMB + lane * 8, qv);
        float sc[CTXW];
        for (int k = 0; k < CTXW; ++k) {
          float kk[8];
          load8f(GW(P.kbuf) + ((size_t)b * CTXW + k) * EMB + lane * 8, kk);
          float s = 0.f;
#pragma unroll
          for (int j = 0; j < 8; ++j) s += qv[j] * kk[j];
          for (int off = 16; off; off >>= 1) s += __shfl_xor(s, off, 32);
          sc[k] = s;
        }
        float mx = sc[0];
        for (int k = 1; k < CTXW; ++k) mx = fmaxf(mx, sc[k]);
        float sum = 0.f;
        for (int k = 0; k < CTXW; ++k) { sc[k] = __expf(sc[k] - mx); sum += sc[k]; }
        const float inv = 1.f / sum;
        float ctx[8] = {0.f, 0.f, 0.f, 0.f, 0.f, 0.f, 0.f, 0.f};
        for (int k = 0; k < CTXW; ++k) {
          float vv[8];
          load8f(GW(P.vbuf) + ((size_t)b * CTXW + k) * EMB + lane * 8, vv);
          const float w = sc[k] * inv;
#pragma unroll
          for (int j = 0; j < 8; ++j) ctx[j] += w * vv[j];
        }
        alignas(16) bf16_t t8[8];
#pragma unroll
        for (int j = 0; j < 8; ++j) t8[j] = (bf16_t)ctx[j];
        *(g_v4i*)(GH(P.din) + (size_t)b * (2 * EMB) + EMB + lane * 8) = *(v4i*)t8;
      } else if (t == CTXW) {        // first decoder step: ctx = current value
        *(g_v4i*)(GH(P.din) + (size_t)b * (2 * EMB) + EMB + lane * 8) =
            *(gc_v4i*)(GW(P.vtmp) + (size_t)b * EMB + lane * 8);
      }
      // append current k,v to ring AFTER attention consumed the old window
      *(g_v4i*)(GH(P.kbuf) + ((size_t)b * CTXW + rs) * EMB + lane * 8) =
          *(gc_v4i*)(GW(P.ktmp) + (size_t)b * EMB + lane * 8);
      *(g_v4i*)(GH(P.vbuf) + ((size_t)b * CTXW + rs) * EMB + lane * 8) =
          *(gc_v4i*)(GW(P.vtmp) + (size_t)b * EMB + lane * 8);
    }
    grid_sync();

    // ------------------------- decoder stack -------------------------------
    if (t >= CTXW) {
      stage(lds,  P.din, BATCH * 2 * EMB, tid);     // [emb | ctx], K=512
      stage(ldsH, hd1r,  BATCH * HID, tid);
      stage_fence();
      lstm_layer<2 * EMB, HID, HID>(lds, GW(P.d1_Wih), ldsH, GW(P.d1_Whh),
                                    GB(P.d1_b), GF(P.c_d1), GH(hd1w), HID,
                                    (g_float*)nullptr, 0, slot, lane);
      grid_sync();

      stage(lds,  hd1w, BATCH * HID, tid);
      stage(ldsH, hd2r, BATCH * HID, tid);
      stage_fence();
      lstm_layer<HID, HID, HID>(lds, GW(P.d2_Wih), ldsH, GW(P.d2_Whh),
                                GB(P.d2_b), GF(P.c_d2), GH(hd2w), HID,
                                (g_float*)nullptr, 0, slot, lane);
      grid_sync();

      stage(lds,  hd2w, BATCH * HID,  tid);
      stage(ldsH, hd3r, BATCH * F_PAD, tid);
      stage_fence();
      lstm_layer<HID, F_PAD, F_IN>(lds, GW(P.d3_Wih), ldsH, GW(P.d3_Whh),
                                   GB(P.d3_b), GF(P.c_d3), GH(hd3w), F_PAD,
                                   GF(P.out), (t - CTXW) * F_IN, slot, lane);
    }
  }
}

// ------------------------------ prep kernels -------------------------------

__global__ void cvt_pad(const float* src, bf16_t* dst, int N, int Ks, int Kd) {
  gc_float* s = (gc_float*)src;
  g_bf16*   d = (g_bf16*)dst;
  const int total = N * Kd;
  for (int i = blockIdx.x * blockDim.x + threadIdx.x; i < total;
       i += gridDim.x * blockDim.x) {
    const int n = i / Kd, k = i - n * Kd;
    d[i] = (bf16_t)((k < Ks) ? s[(size_t)n * Ks + k] : 0.f);
  }
}

__global__ void init_ws(bf16_t* bz, size_t bn, float* fz, size_t fn) {
  g_bf16*  b = (g_bf16*)bz;
  g_float* f = (g_float*)fz;
  const size_t stride = (size_t)gridDim.x * blockDim.x;
  for (size_t i = blockIdx.x * (size_t)blockDim.x + threadIdx.x; i < bn; i += stride)
    b[i] = (bf16_t)0.f;
  for (size_t i = blockIdx.x * (size_t)blockDim.x + threadIdx.x; i < fn; i += stride)
    f[i] = 0.f;
}

// --------------------------------- host ------------------------------------

extern "C" void kernel_launch(void* const* d_in, const int* in_sizes, int n_in,
                              void* d_out, int out_size, void* d_ws, size_t ws_size,
                              hipStream_t stream) {
  (void)in_sizes; (void)n_in; (void)out_size; (void)ws_size;

  bf16_t* base = (bf16_t*)d_ws;
  size_t off = 0;
  auto alloc = [&](size_t n) { bf16_t* r = base + off; off += n; return r; };

  // bf16 weight region (padded, row-major [N][Kpad])
  bf16_t* w_emb    = alloc(256 * 96);
  bf16_t* w_e1_wih = alloc(1024 * 96);
  bf16_t* w_e1_whh = alloc(1024 * 256);
  bf16_t* w_e2_wih = alloc(1024 * 256);
  bf16_t* w_e2_whh = alloc(1024 * 256);
  bf16_t* w_e3_wih = alloc(1024 * 256);
  bf16_t* w_e3_whh = alloc(1024 * 256);
  bf16_t* w_d1_wih = alloc(1024 * 512);
  bf16_t* w_d1_whh = alloc(1024 * 256);
  bf16_t* w_d2_wih = alloc(1024 * 256);
  bf16_t* w_d2_whh = alloc(1024 * 256);
  bf16_t* w_d3_wih = alloc(320 * 256);
  bf16_t* w_d3_whh = alloc(320 * 96);
  bf16_t* w_key    = alloc(256 * 256);
  bf16_t* w_val    = alloc(256 * 256);

  // bf16 activation/state region (zeroed every call)
  const size_t actStart = off;
  SeqParams P;
  P.xt    = alloc(64 * 96);
  P.xprev = alloc(64 * 96);
  P.din   = alloc(64 * 512);
  P.ktmp  = alloc(64 * 256);
  P.vtmp  = alloc(64 * 256);
  P.kbuf  = alloc(64 * 20 * 256);
  P.vbuf  = alloc(64 * 20 * 256);
  for (int i = 0; i < 2; ++i) P.h_e1[i] = alloc(64 * 256);
  for (int i = 0; i < 2; ++i) P.h_e2[i] = alloc(64 * 256);
  for (int i = 0; i < 2; ++i) P.h_e3[i] = alloc(64 * 256);
  for (int i = 0; i < 2; ++i) P.h_d1[i] = alloc(64 * 256);
  for (int i = 0; i < 2; ++i) P.h_d2[i] = alloc(64 * 256);
  for (int i = 0; i < 2; ++i) P.h_d3[i] = alloc(64 * 96);
  const size_t actCount = off - actStart;

  off = (off + 1) & ~(size_t)1;             // 4B-align fp32 region
  float* fbase = (float*)(base + off);
  P.c_e1 = fbase;
  P.c_e2 = P.c_e1 + 64 * 256;
  P.c_e3 = P.c_e2 + 64 * 256;
  P.c_d1 = P.c_e3 + 64 * 256;
  P.c_d2 = P.c_d1 + 64 * 256;
  P.c_d3 = P.c_d2 + 64 * 256;
  const size_t fCount = 5 * 64 * 256 + 64 * 80;

  // inputs (setup_inputs order)
  P.x     = (const float*)d_in[0];
  P.emb_b = (const float*)d_in[2];
  P.e1_b  = (const float*)d_in[5];
  P.e2_b  = (const float*)d_in[8];
  P.e3_b  = (const float*)d_in[11];
  P.d1_b  = (const float*)d_in[14];
  P.d2_b  = (const float*)d_in[17];
  P.d3_b  = (const float*)d_in[20];
  P.key_b = (const float*)d_in[22];
  P.val_b = (const float*)d_in[24];
  P.emb_W = w_emb;    P.e1_Wih = w_e1_wih; P.e1_Whh = w_e1_whh;
  P.e2_Wih = w_e2_wih; P.e2_Whh = w_e2_whh;
  P.e3_Wih = w_e3_wih; P.e3_Whh = w_e3_whh;
  P.d1_Wih = w_d1_wih; P.d1_Whh = w_d1_whh;
  P.d2_Wih = w_d2_wih; P.d2_Whh = w_d2_whh;
  P.d3_Wih = w_d3_wih; P.d3_Whh = w_d3_whh;
  P.key_W = w_key;    P.val_W = w_val;
  P.out   = (float*)d_out;

  auto cvt = [&](int idx, bf16_t* dst, int N, int Ks, int Kd) {
    int blocks = (N * Kd + 255) / 256;
    if (blocks > 2048) blocks = 2048;
    cvt_pad<<<dim3(blocks), dim3(256), 0, stream>>>((const float*)d_in[idx],
                                                    dst, N, Ks, Kd);
  };
  cvt(1,  w_emb,    256,  80,  96);
  cvt(3,  w_e1_wih, 1024, 80,  96);
  cvt(4,  w_e1_whh, 1024, 256, 256);
  cvt(6,  w_e2_wih, 1024, 256, 256);
  cvt(7,  w_e2_whh, 1024, 256, 256);
  cvt(9,  w_e3_wih, 1024, 256, 256);
  cvt(10, w_e3_whh, 1024, 256, 256);
  cvt(12, w_d1_wih, 1024, 512, 512);
  cvt(13, w_d1_whh, 1024, 256, 256);
  cvt(15, w_d2_wih, 1024, 256, 256);
  cvt(16, w_d2_whh, 1024, 256, 256);
  cvt(18, w_d3_wih, 320,  256, 256);
  cvt(19, w_d3_whh, 320,  80,  96);
  cvt(21, w_key,    256,  256, 256);
  cvt(23, w_val,    256,  256, 256);

  init_ws<<<dim3(512), dim3(256), 0, stream>>>(base + actStart, actCount,
                                               fbase, fCount);

  seq2seq_main<<<dim3(NWG), dim3(THREADS), LDS_BYTES, stream>>>(P);
}
